// QuickCFGenerator_22385369546871
// MI455X (gfx1250) — compile-verified, matrix operand order checked by reading, not run on hardware
//
#include <hip/hip_runtime.h>

typedef __attribute__((ext_vector_type(16))) _Float16 v16h;
typedef __attribute__((ext_vector_type(8)))  _Float16 v8h;
typedef __attribute__((ext_vector_type(4)))  _Float16 v4h;
typedef __attribute__((ext_vector_type(8)))  float    v8f;

#define NHEAD  8
#define DH     64
#define DMODEL 512
#define FFDIM  2048
#define LAYERS 4
#define BB     4
#define PP     1024
#define LL     2048
#define ROWS   (BB * LL)   // 8192

// Build a 16-half WMMA fragment from two 16-byte LDS chunks.
__device__ __forceinline__ v16h frag_ld(const _Float16* p, int lo, int hi) {
  v8h a = *(const v8h*)(p + lo);
  v8h b = *(const v8h*)(p + hi);
  return __builtin_shufflevector(a, b, 0,1,2,3,4,5,6,7,8,9,10,11,12,13,14,15);
}

__device__ __forceinline__ void wait_ds0() {
  asm volatile("s_wait_dscnt 0" ::: "memory");
}

// Scheduling groups: issue NDS DS-reads, then NWMMA WMMAs, back-to-back.
// Masks: 0x100 = DS-read, 0x008 = MFMA/WMMA. (Args must be ICEs -> template.)
template<int NDS, int NWMMA>
__device__ __forceinline__ void sched_ds_then_wmma() {
  __builtin_amdgcn_sched_group_barrier(0x100, NDS, 0);
  __builtin_amdgcn_sched_group_barrier(0x008, NWMMA, 0);
}

// DPP16 row_ror:N move (rows of 16 lanes) — pure-VALU cross-lane reduction.
template<int CTRL>
__device__ __forceinline__ float dpp_f(float x) {
  return __int_as_float(
      __builtin_amdgcn_update_dpp(0, __float_as_int(x), CTRL, 0xF, 0xF, true));
}
__device__ __forceinline__ float rowmax16(float x) {
  x = fmaxf(x, dpp_f<0x121>(x));   // row_ror:1
  x = fmaxf(x, dpp_f<0x122>(x));   // row_ror:2
  x = fmaxf(x, dpp_f<0x124>(x));   // row_ror:4
  x = fmaxf(x, dpp_f<0x128>(x));   // row_ror:8
  return x;
}
__device__ __forceinline__ float rowsum16(float x) {
  x += dpp_f<0x121>(x);
  x += dpp_f<0x122>(x);
  x += dpp_f<0x124>(x);
  x += dpp_f<0x128>(x);
  return x;
}

// ---------------------------------------------------------------------------
// C[M,N] = act(A[M,K] @ W[K,N] + bias[N]) ; f16 WMMA, f32 accumulate.
// Block: 256 thr = 8 waves. Tile 128x128x32. Wave tile 32x64 (2x4 wmma).
// ---------------------------------------------------------------------------
template<int ACT>
__global__ __launch_bounds__(256)
void gemm_bias(const float* __restrict__ A, const float* __restrict__ W,
               const float* __restrict__ bias, float* __restrict__ C,
               int M, int N, int K) {
  __shared__ __align__(16) _Float16 As[128 * 32];   // row-major (m, k)
  __shared__ __align__(16) _Float16 Bs[128 * 32];   // n-major   (n, k)
  const int tid  = threadIdx.x;
  const int lane = tid & 31, wid = tid >> 5;
  const int wm = wid >> 1, wn = wid & 1;           // 4 x 2 wave grid
  const int row0 = blockIdx.y * 128, col0 = blockIdx.x * 128;
  const int l15  = lane & 15;
  const int half = (lane < 16) ? 0 : 8;

  v8f acc[2][4];
#pragma unroll
  for (int i = 0; i < 2; ++i)
#pragma unroll
    for (int j = 0; j < 4; ++j) acc[i][j] = (v8f)0.0f;

  for (int k0 = 0; k0 < K; k0 += 32) {
    // Load A tile (coalesced), convert f32->f16, contiguous LDS store.
#pragma unroll
    for (int r = 0; r < 4; ++r) {
      int flat = r * 1024 + tid * 4;
      int m = flat >> 5, kk = flat & 31;
      float4 va = *(const float4*)(A + (size_t)(row0 + m) * K + k0 + kk);
      v4h h4; h4[0] = (_Float16)va.x; h4[1] = (_Float16)va.y;
      h4[2] = (_Float16)va.z; h4[3] = (_Float16)va.w;
      *(v4h*)(As + m * 32 + kk) = h4;
    }
    // Load W tile (coalesced), transpose into n-major LDS.
#pragma unroll
    for (int r = 0; r < 4; ++r) {
      int flat = r * 1024 + tid * 4;
      int kk = flat >> 7, n = flat & 127;
      float4 vb = *(const float4*)(W + (size_t)(k0 + kk) * N + col0 + n);
      Bs[(n + 0) * 32 + kk] = (_Float16)vb.x;
      Bs[(n + 1) * 32 + kk] = (_Float16)vb.y;
      Bs[(n + 2) * 32 + kk] = (_Float16)vb.z;
      Bs[(n + 3) * 32 + kk] = (_Float16)vb.w;
    }
    // Prefetch next K-step's tiles into cache while this tile computes.
    if (k0 + 32 < K) {
      __builtin_prefetch(A + (size_t)(row0 + (tid >> 1)) * K + k0 + 32 +
                             (tid & 1) * 16, 0, 1);
      __builtin_prefetch(W + (size_t)(k0 + 32 + (tid >> 3)) * N + col0 +
                             (tid & 7) * 16, 0, 1);
    }
    __syncthreads();

    v16h af[2], bf[4];
#pragma unroll
    for (int i = 0; i < 2; ++i) {
      int m = wm * 32 + i * 16 + l15;
      af[i] = frag_ld(As, m * 32 + half, m * 32 + 16 + half);
    }
#pragma unroll
    for (int j = 0; j < 4; ++j) {
      int n = wn * 64 + j * 16 + l15;
      bf[j] = frag_ld(Bs, n * 32 + half, n * 32 + 16 + half);
    }
#pragma unroll
    for (int i = 0; i < 2; ++i)
#pragma unroll
      for (int j = 0; j < 4; ++j)
        acc[i][j] = __builtin_amdgcn_wmma_f32_16x16x32_f16(
            false, af[i], false, bf[j], (short)0, acc[i][j], false, false);
    sched_ds_then_wmma<12, 8>();   // 12 ds_load_b128, then 8 v_wmma
    __syncthreads();
  }

#pragma unroll
  for (int i = 0; i < 2; ++i) {
#pragma unroll
    for (int j = 0; j < 4; ++j) {
      int nn = col0 + wn * 64 + j * 16 + l15;
      float bv = bias[nn];
#pragma unroll
      for (int r = 0; r < 8; ++r) {
        int m = row0 + wm * 32 + i * 16 + ((lane < 16) ? r : 8 + r);
        float v = acc[i][j][r] + bv;
        if (ACT) v = fmaxf(v, 0.0f);
        C[(size_t)m * N + nn] = v;
      }
    }
  }
}

// ---------------------------------------------------------------------------
// Sparse attention (flash-style online softmax over the P-key prefix,
// plus a per-row self-attention diagonal term for "gen" queries).
// Block = (b,h) x 64 query rows; 4 waves x 16 rows. dh = 64. scale = 0.125.
// qkv layout: [B, L, 3, H, dh]  (row stride 1536 floats).
// out layout: [B, L, H, dh] == [B*L, 512].
// ---------------------------------------------------------------------------
__global__ __launch_bounds__(128)
void sparse_attention(const float* __restrict__ qkv, float* __restrict__ out) {
  __shared__ __align__(16) _Float16 Qs[64 * 64];
  __shared__ __align__(16) _Float16 Ks[32 * 64];
  __shared__ __align__(16) _Float16 Vt[64 * 32];   // (dh, key)
  __shared__ __align__(16) _Float16 Ps[4 * 16 * 32];
  __shared__ float Mrow[64], Lrow[64], Frow[64], Erow[64];

  const int tid = threadIdx.x, lane = tid & 31, wid = tid >> 5;
  const int b = blockIdx.y / NHEAD, h = blockIdx.y % NHEAD;
  const int row0 = blockIdx.x * 64;
  const bool gen = (row0 >= PP);
  const int l15 = lane & 15;
  const int half = (lane < 16) ? 0 : 8;
  const int wr0 = wid * 16;

  const float* qptr = qkv + ((size_t)(b * LL) * 3) * DMODEL + h * DH;

  // Load 64x64 Q tile into LDS.
#pragma unroll
  for (int it = 0; it < 8; ++it) {
    int flat = it * 512 + tid * 4;
    int m = flat >> 6, dd = flat & 63;
    float4 v4 = *(const float4*)(qptr + (size_t)(row0 + m) * 1536 + dd);
    v4h h4; h4[0] = (_Float16)v4.x; h4[1] = (_Float16)v4.y;
    h4[2] = (_Float16)v4.z; h4[3] = (_Float16)v4.w;
    *(v4h*)(Qs + m * 64 + dd) = h4;
  }
  __syncthreads();

  v16h qa[2];
#pragma unroll
  for (int s = 0; s < 2; ++s) {
    int m = wr0 + l15;
    qa[s] = frag_ld(Qs, m * 64 + s * 32 + half, m * 64 + s * 32 + 16 + half);
  }

  float mr[8], lr[8];
  v8f acco[4];
#pragma unroll
  for (int r = 0; r < 8; ++r) { mr[r] = -1.0e30f; lr[r] = 0.0f; }
#pragma unroll
  for (int j = 0; j < 4; ++j) acco[j] = (v8f)0.0f;

  for (int c = 0; c < PP / 32; ++c) {
    const int k0 = c * 32;
    __syncthreads();   // previous chunk's K/V reads done
#pragma unroll
    for (int it = 0; it < 4; ++it) {
      int flat = it * 512 + tid * 4;
      int kk = flat >> 6, dd = flat & 63;
      float4 vk = *(const float4*)(qptr + 512 + (size_t)(k0 + kk) * 1536 + dd);
      v4h h4; h4[0] = (_Float16)vk.x; h4[1] = (_Float16)vk.y;
      h4[2] = (_Float16)vk.z; h4[3] = (_Float16)vk.w;
      *(v4h*)(Ks + kk * 64 + dd) = h4;
      float4 vv = *(const float4*)(qptr + 1024 + (size_t)(k0 + kk) * 1536 + dd);
      Vt[(dd + 0) * 32 + kk] = (_Float16)vv.x;
      Vt[(dd + 1) * 32 + kk] = (_Float16)vv.y;
      Vt[(dd + 2) * 32 + kk] = (_Float16)vv.z;
      Vt[(dd + 3) * 32 + kk] = (_Float16)vv.w;
    }
    // Prefetch next chunk's K/V rows (global_prefetch_b8).
    if (k0 + 32 < PP) {
      const float* nkr = qptr + 512 +
          (size_t)(k0 + 32 + (tid >> 2)) * 1536 + (tid & 3) * 16;
      __builtin_prefetch(nkr, 0, 1);
      __builtin_prefetch(nkr + 512, 0, 1);
    }
    __syncthreads();

    // S = (Q Kp^T) * scale : two 16x16 key sub-tiles, K-dim = dh = 64.
    v8f st[2];
#pragma unroll
    for (int j = 0; j < 2; ++j) {
      v8f z = (v8f)0.0f;
      int n = j * 16 + l15;
      v16h kb0 = frag_ld(Ks, n * 64 + half,      n * 64 + 16 + half);
      v16h kb1 = frag_ld(Ks, n * 64 + 32 + half, n * 64 + 48 + half);
      z = __builtin_amdgcn_wmma_f32_16x16x32_f16(
          false, qa[0], false, kb0, (short)0, z, false, false);
      z = __builtin_amdgcn_wmma_f32_16x16x32_f16(
          false, qa[1], false, kb1, (short)0, z, false, false);
      st[j] = z * 0.125f;
    }
    sched_ds_then_wmma<8, 4>();

    // Online softmax; row stats via DPP16 row_ror reductions (pure VALU).
#pragma unroll
    for (int r = 0; r < 8; ++r) {
      float mx = rowmax16(fmaxf(st[0][r], st[1][r]));
      float nm = fmaxf(mr[r], mx);
      float fs = __expf(mr[r] - nm);
      float p0 = __expf(st[0][r] - nm);
      float p1 = __expf(st[1][r] - nm);
      float rs = rowsum16(p0 + p1);
      lr[r] = lr[r] * fs + rs;
      mr[r] = nm;
#pragma unroll
      for (int jj = 0; jj < 4; ++jj) acco[jj][r] *= fs;
      int rl = (lane < 16) ? r : 8 + r;
      Ps[wid * 512 + rl * 32 + l15]      = (_Float16)p0;
      Ps[wid * 512 + rl * 32 + 16 + l15] = (_Float16)p1;
    }
    wait_ds0();   // wave-local: P staging stores visible before fragment reload

    v16h pf = frag_ld(Ps + wid * 512, l15 * 32 + half, l15 * 32 + 16 + half);
    v16h vf[4];
#pragma unroll
    for (int jj = 0; jj < 4; ++jj) {
      int n = jj * 16 + l15;
      vf[jj] = frag_ld(Vt, n * 32 + half, n * 32 + 16 + half);
    }
#pragma unroll
    for (int jj = 0; jj < 4; ++jj)
      acco[jj] = __builtin_amdgcn_wmma_f32_16x16x32_f16(
          false, pf, false, vf[jj], (short)0, acco[jj], false, false);
    sched_ds_then_wmma<10, 4>();
  }

  // Dump per-row stats (lane 0 -> rows r, lane 16 -> rows 8+r).
  if (l15 == 0) {
    int rb = wr0 + (lane >> 4) * 8;
#pragma unroll
    for (int r = 0; r < 8; ++r) { Mrow[rb + r] = mr[r]; Lrow[rb + r] = lr[r]; }
  }
  __syncthreads();

  const int rown = wr0 + l15;
  if (gen) {
    // dlog = (q . k_self) * scale ; lanes L / L+16 split the 64-dim dot.
    int gr = row0 + rown;
    const float* qr = qptr + (size_t)gr * 1536 + (lane >> 4) * 32;
    const float* kr = qr + 512;
    float part = 0.0f;
#pragma unroll
    for (int i = 0; i < 32; i += 4) {
      float4 a4 = *(const float4*)(qr + i);
      float4 b4 = *(const float4*)(kr + i);
      part += a4.x * b4.x + a4.y * b4.y + a4.z * b4.z + a4.w * b4.w;
    }
    part += __shfl_xor(part, 16, 32);   // combine the two 32-elem halves
    float dlog = part * 0.125f;
    float mo = Mrow[rown], lo2 = Lrow[rown];
    float nm = fmaxf(mo, dlog);
    float fs = __expf(mo - nm);
    float ed = __expf(dlog - nm);
    if (lane < 16) {
      Lrow[rown] = lo2 * fs + ed;
      Frow[rown] = fs;
      Erow[rown] = ed;
    }
  } else if (lane < 16) {
    Frow[rown] = 1.0f;
    Erow[rown] = 0.0f;
  }
  __syncthreads();

  float* optr = out + (size_t)(b * LL) * DMODEL + h * DH;
#pragma unroll
  for (int jj = 0; jj < 4; ++jj) {
    int col = jj * 16 + l15;
#pragma unroll
    for (int r = 0; r < 8; ++r) {
      int rl = wr0 + ((lane < 16) ? r : 8 + r);
      int gr = row0 + rl;
      float v = acco[jj][r] * Frow[rl];
      if (gen) v += Erow[rl] * qptr[1024 + (size_t)gr * 1536 + col];
      v /= Lrow[rl];
      optr[(size_t)gr * DMODEL + col] = v;
    }
  }
}

// ---------------------------------------------------------------------------
// y = LayerNorm(x + a) * g + beta ; one block per row (d = 512), in-place ok.
// ---------------------------------------------------------------------------
__global__ __launch_bounds__(256)
void add_ln(const float* __restrict__ x, const float* __restrict__ a,
            const float* __restrict__ g, const float* __restrict__ be,
            float* __restrict__ y) {
  __shared__ float red[256];
  const int row = blockIdx.x, t = threadIdx.x;
  const float* xr = x + (size_t)row * DMODEL;
  const float* ar = a + (size_t)row * DMODEL;
  float v0 = xr[t] + ar[t];
  float v1 = xr[t + 256] + ar[t + 256];
  red[t] = v0 + v1;
  __syncthreads();
#pragma unroll
  for (int off = 128; off > 0; off >>= 1) {
    if (t < off) red[t] += red[t + off];
    __syncthreads();
  }
  float mu = red[0] * (1.0f / 512.0f);
  __syncthreads();
  float d0 = v0 - mu, d1 = v1 - mu;
  red[t] = d0 * d0 + d1 * d1;
  __syncthreads();
#pragma unroll
  for (int off = 128; off > 0; off >>= 1) {
    if (t < off) red[t] += red[t + off];
    __syncthreads();
  }
  float inv = rsqrtf(red[0] * (1.0f / 512.0f) + 1e-5f);
  y[(size_t)row * DMODEL + t]       = d0 * inv * g[t] + be[t];
  y[(size_t)row * DMODEL + t + 256] = d1 * inv * g[t + 256] + be[t + 256];
}

// ---------------------------------------------------------------------------
__global__ __launch_bounds__(256)
void pack_x(const float* __restrict__ p, const float* __restrict__ gn,
            float* __restrict__ X) {
  size_t idx = (size_t)blockIdx.x * 256 + threadIdx.x;
  int c = (int)(idx % DMODEL);
  int l = (int)((idx / DMODEL) % LL);
  int b = (int)(idx / ((size_t)DMODEL * LL));
  float v = (l < PP) ? p[((size_t)b * PP + l) * DMODEL + c]
                     : gn[((size_t)b * PP + (l - PP)) * DMODEL + c];
  X[idx] = v;
}

__global__ __launch_bounds__(256)
void unpack_x(const float* __restrict__ X, float* __restrict__ out) {
  size_t idx = (size_t)blockIdx.x * 256 + threadIdx.x;
  const size_t halfN = (size_t)BB * PP * DMODEL;
  size_t i = idx;
  int l;
  if (i < halfN) {
    int b = (int)(i / ((size_t)PP * DMODEL));
    size_t rem = i % ((size_t)PP * DMODEL);
    l = (int)(rem / DMODEL);
    int c = (int)(rem % DMODEL);
    out[idx] = X[((size_t)b * LL + l) * DMODEL + c];
  } else {
    i -= halfN;
    int b = (int)(i / ((size_t)PP * DMODEL));
    size_t rem = i % ((size_t)PP * DMODEL);
    l = PP + (int)(rem / DMODEL);
    int c = (int)(rem % DMODEL);
    out[idx] = X[((size_t)b * LL + l) * DMODEL + c];
  }
}

// ---------------------------------------------------------------------------
extern "C" void kernel_launch(void* const* d_in, const int* in_sizes, int n_in,
                              void* d_out, int out_size, void* d_ws, size_t ws_size,
                              hipStream_t stream) {
  const float* pcpt = (const float*)d_in[0];
  const float* genp = (const float*)d_in[1];
  const float* Wqkv = (const float*)d_in[3];
  const float* bqkv = (const float*)d_in[4];
  const float* Wo   = (const float*)d_in[5];
  const float* bo   = (const float*)d_in[6];
  const float* W1   = (const float*)d_in[7];
  const float* b1   = (const float*)d_in[8];
  const float* W2   = (const float*)d_in[9];
  const float* b2   = (const float*)d_in[10];
  const float* g1   = (const float*)d_in[11];
  const float* be1  = (const float*)d_in[12];
  const float* g2   = (const float*)d_in[13];
  const float* be2  = (const float*)d_in[14];

  float* X   = (float*)d_ws;                         // 8192 x 512
  float* QKV = X + (size_t)ROWS * DMODEL;            // 8192 x 1536
  float* ATT = QKV + (size_t)ROWS * 3 * DMODEL;      // 8192 x 512
  float* FF2 = ATT + (size_t)ROWS * DMODEL;          // 8192 x 512
  float* BUF = QKV;                                  // proj-out (QKV dead)
  float* TMP = QKV;                                  // ffn hidden (QKV+ATT span)

  pack_x<<<dim3((ROWS * DMODEL) / 256), 256, 0, stream>>>(pcpt, genp, X);

  for (int i = 0; i < LAYERS; ++i) {
    gemm_bias<0><<<dim3(12, 64), 256, 0, stream>>>(
        X, Wqkv + (size_t)i * DMODEL * 3 * DMODEL, bqkv + (size_t)i * 3 * DMODEL,
        QKV, ROWS, 3 * DMODEL, DMODEL);
    sparse_attention<<<dim3(LL / 64, BB * NHEAD), 128, 0, stream>>>(QKV, ATT);
    gemm_bias<0><<<dim3(4, 64), 256, 0, stream>>>(
        ATT, Wo + (size_t)i * DMODEL * DMODEL, bo + (size_t)i * DMODEL,
        BUF, ROWS, DMODEL, DMODEL);
    add_ln<<<dim3(ROWS), 256, 0, stream>>>(
        X, BUF, g1 + (size_t)i * DMODEL, be1 + (size_t)i * DMODEL, X);
    gemm_bias<1><<<dim3(16, 64), 256, 0, stream>>>(
        X, W1 + (size_t)i * DMODEL * FFDIM, b1 + (size_t)i * FFDIM,
        TMP, ROWS, FFDIM, DMODEL);
    gemm_bias<0><<<dim3(4, 64), 256, 0, stream>>>(
        TMP, W2 + (size_t)i * FFDIM * DMODEL, b2 + (size_t)i * DMODEL,
        FF2, ROWS, DMODEL, FFDIM);
    add_ln<<<dim3(ROWS), 256, 0, stream>>>(
        X, FF2, g2 + (size_t)i * DMODEL, be2 + (size_t)i * DMODEL, X);
  }

  unpack_x<<<dim3((ROWS * DMODEL) / 256), 256, 0, stream>>>(X, (float*)d_out);
}
